// GatedDirGCNConv_19439021982026
// MI455X (gfx1250) — matrix operand order repaired and checked by reading, not compiled, not run on hardware
//
#include <hip/hip_runtime.h>

// ---------------------------------------------------------------------------
// GatedDirGCNConv for MI455X (gfx1250, wave32, WMMA f32_16x16x32_f16).
//
// Workspace layout (floats):  h_in[N*128] | h_out[N*128] | in_deg[N] |
//                             out_deg[N]  | scores[E]
// Node arrays (~77MB) are L2-resident (192MB L2); weights live in LDS
// pre-swizzled into WMMA B-fragment layout. Gather latency is hidden by
// (a) batching A-fragment loads and (b) global_prefetch_b8 of the NEXT
// group's rows into the WGP cache one iteration ahead.
// ---------------------------------------------------------------------------

typedef __attribute__((ext_vector_type(16))) _Float16 v16h;
typedef __attribute__((ext_vector_type(8)))  float    v8f;
typedef __attribute__((ext_vector_type(4)))  float    f4;

#define WPB 8   // waves per 256-thread block

#define WMMA_F16(a, b, c) \
  __builtin_amdgcn_wmma_f32_16x16x32_f16(false, (a), false, (b), (short)0, (c), false, false)

// Native CDNA5 FP32 global atomic add (GLOBAL_ATOMIC_ADD_F32, no-return form).
__device__ __forceinline__ void atomic_add_f32(float* p, float v) {
  asm volatile("global_atomic_add_f32 %0, %1, off" : : "v"(p), "v"(v) : "memory");
}

// Prefetch one 512B node-feature row (4 x 128B) into near caches.
__device__ __forceinline__ void prefetch_row(const float* p) {
  __builtin_prefetch(p, 0, 1);
  __builtin_prefetch(p + 32, 0, 1);
  __builtin_prefetch(p + 64, 0, 1);
  __builtin_prefetch(p + 96, 0, 1);
}

// Stage a K x 128 fp32 weight matrix into LDS as f16, pre-swizzled into the
// WMMA B-fragment layout for 16x16x32 tiles:
//   B (32x16) fragment: lane<16 holds col N=lane,    K = kt*32 + {0..7, 16..23}
//                       lane>=16 holds col N=lane-16, K = kt*32 + {8..15, 24..31}
// Storage: tile (kt,nt) -> 32 lanes x 16 halfs contiguous (32B per lane).
__device__ __forceinline__ void stage_w(const float* __restrict__ W,
                                        _Float16* lw, int KT) {
  int total = KT * 8 * 32 * 16;
  for (int i = threadIdx.x; i < total; i += blockDim.x) {
    int j    = i & 15;
    int lane = (i >> 4) & 31;
    int tile = i >> 9;
    int kt = tile >> 3, nt = tile & 7;
    int k = kt * 32 + ((lane & 16) >> 1) + (j & 7) + ((j & 8) << 1);
    int n = nt * 16 + (lane & 15);
    lw[i] = (_Float16)W[k * 128 + n];
  }
}

// Build a 16x32 f16 A-fragment for one row of a 128-wide fp32 matrix.
// lane<16: row M=lane&15, K = kbase + {0..7, 16..23}
// lane>=16: same rows,    K = kbase + {8..15, 24..31}   (hiOff = 8)
__device__ __forceinline__ v16h make_a_frag(const float* __restrict__ rowp,
                                            int kbase, int hiOff, float scale) {
  const float* p = rowp + kbase + hiOff;
  f4 a0 = *(const f4*)(p);
  f4 a1 = *(const f4*)(p + 4);
  f4 b0 = *(const f4*)(p + 16);
  f4 b1 = *(const f4*)(p + 20);
  v16h r;
  r[0]  = (_Float16)(a0.x * scale); r[1]  = (_Float16)(a0.y * scale);
  r[2]  = (_Float16)(a0.z * scale); r[3]  = (_Float16)(a0.w * scale);
  r[4]  = (_Float16)(a1.x * scale); r[5]  = (_Float16)(a1.y * scale);
  r[6]  = (_Float16)(a1.z * scale); r[7]  = (_Float16)(a1.w * scale);
  r[8]  = (_Float16)(b0.x * scale); r[9]  = (_Float16)(b0.y * scale);
  r[10] = (_Float16)(b0.z * scale); r[11] = (_Float16)(b0.w * scale);
  r[12] = (_Float16)(b1.x * scale); r[13] = (_Float16)(b1.y * scale);
  r[14] = (_Float16)(b1.z * scale); r[15] = (_Float16)(b1.w * scale);
  return r;
}

__device__ __forceinline__ v16h load_b_frag(const _Float16* lw, int tile, int lane) {
  return *(const v16h*)(lw + ((tile * 32 + lane) << 4));
}

__device__ __forceinline__ float sigmoidf(float z) {
  return 1.0f / (1.0f + __expf(-z));
}

// --------------------------- kernel 0: zero ws ------------------------------
__global__ void zero_kernel(float* __restrict__ p, long n4) {
  long i = (long)blockIdx.x * blockDim.x + threadIdx.x;
  long stride = (long)gridDim.x * blockDim.x;
  f4 z = {};
  for (; i < n4; i += stride) ((f4*)p)[i] = z;
}

// ------------------- kernel 1: edge MLP -> score + degrees ------------------
__global__ __launch_bounds__(256)
void edge_score_kernel(const float* __restrict__ x,
                       const long long* __restrict__ ei,   // [2,E] int64
                       const float* __restrict__ w_e1, const float* __restrict__ b_e1,
                       const float* __restrict__ w_e2, const float* __restrict__ b_e2,
                       float* __restrict__ scores,
                       float* __restrict__ in_deg, float* __restrict__ out_deg,
                       int E_) {
  __shared__ __align__(32) _Float16 lw[8 * 8 * 32 * 16];   // 64KB: w_e1 (256x128)
  stage_w(w_e1, lw, 8);
  __syncthreads();

  const int lane = threadIdx.x & 31;
  const int wid  = threadIdx.x >> 5;
  const int lm   = lane & 15;
  const int hiOff = (lane & 16) >> 1;            // 0 or 8
  const bool hi  = lane >= 16;
  const int nGroups = E_ >> 4;
  const int stride  = gridDim.x * WPB;
  const float be2 = b_e2[0];

  // per-lane epilogue constants (group-invariant): column n = nt*16 + lm
  float biasv[8], w2v[8];
#pragma unroll
  for (int nt = 0; nt < 8; nt++) {
    biasv[nt] = b_e1[nt * 16 + lm];
    w2v[nt]   = w_e2[nt * 16 + lm];
  }

  for (int g = blockIdx.x * WPB + wid; g < nGroups; g += stride) {
    const int e0 = g << 4;

    // ---- prefetch next group's 32 rows into near cache (one iter ahead) ----
    {
      int gn = g + stride;
      if (gn < nGroups) {
        int en = gn << 4;
        int pn = 0;
        if (lane < 16) pn = (int)ei[(hi ? E_ : 0) + en + lane];  // lanes<16 only
        // lanes 0-15 -> next src rows, lanes 16-31 -> next dst rows
        int sv = __shfl(pn, lm, 32);
        int dv = 0;
        if (lane < 16) dv = (int)ei[E_ + en + lane];
        dv = __shfl(dv, lm, 32);
        prefetch_row(x + (long)(hi ? dv : sv) * 128);
      }
    }

    int s_i = 0, d_i = 0;
    if (lane < 16) {
      s_i = (int)ei[e0 + lane];
      d_i = (int)ei[E_ + e0 + lane];
      atomic_add_f32(out_deg + s_i, 1.0f);
      atomic_add_f32(in_deg + d_i, 1.0f);
    }
    const int srcm = __shfl(s_i, lm, 32);
    const int dstm = __shfl(d_i, lm, 32);
    const float* ps = x + (long)srcm * 128;
    const float* pd = x + (long)dstm * 128;

    // issue the A-fragment gathers (up to 32 global_load_b128 in flight)
    v16h af[8];
#pragma unroll
    for (int kt = 0; kt < 4; kt++) af[kt]     = make_a_frag(ps, kt * 32, hiOff, 1.0f);
#pragma unroll
    for (int kt = 0; kt < 4; kt++) af[kt + 4] = make_a_frag(pd, kt * 32, hiOff, 1.0f);

    v8f acc[8];
#pragma unroll
    for (int t = 0; t < 8; t++) { v8f z = {}; acc[t] = z; }

#pragma unroll
    for (int kt = 0; kt < 8; kt++)
#pragma unroll
      for (int nt = 0; nt < 8; nt++)
        acc[nt] = WMMA_F16(af[kt], load_b_frag(lw, kt * 8 + nt, lane), acc[nt]);

    // bias + ReLU + dot with w_e2 (per-row partial over this lane's columns)
    float part[8];
#pragma unroll
    for (int r = 0; r < 8; r++) part[r] = 0.0f;
#pragma unroll
    for (int nt = 0; nt < 8; nt++)
#pragma unroll
      for (int r = 0; r < 8; r++) {
        float h = acc[nt][r] + biasv[nt];
        h = h > 0.0f ? h : 0.0f;
        part[r] += h * w2v[nt];
      }
    // reduce across each 16-lane half (rows 0-7 in lanes 0-15, rows 8-15 in 16-31)
#pragma unroll
    for (int off = 1; off < 16; off <<= 1)
#pragma unroll
      for (int r = 0; r < 8; r++) part[r] += __shfl_xor(part[r], off, 32);

    if (lm == 0) {
#pragma unroll
      for (int r = 0; r < 8; r++)
        scores[e0 + (hi ? r + 8 : r)] = sigmoidf(part[r] + be2);
    }
  }
}

// -------------- kernel 2: directional messages + scatter-add ----------------
__global__ __launch_bounds__(256)
void edge_msg_kernel(const float* __restrict__ x,
                     const long long* __restrict__ ei,
                     const float* __restrict__ w_s2d, const float* __restrict__ b_s2d,
                     const float* __restrict__ w_d2s, const float* __restrict__ b_d2s,
                     const float* __restrict__ scores,
                     float* __restrict__ h_in, float* __restrict__ h_out,
                     int E_) {
  __shared__ __align__(32) _Float16 lwA[4 * 8 * 32 * 16];  // 32KB w_s2d
  __shared__ __align__(32) _Float16 lwB[4 * 8 * 32 * 16];  // 32KB w_d2s
  stage_w(w_s2d, lwA, 4);
  stage_w(w_d2s, lwB, 4);
  __syncthreads();

  const int lane = threadIdx.x & 31;
  const int wid  = threadIdx.x >> 5;
  const int lm   = lane & 15;
  const int hiOff = (lane & 16) >> 1;
  const bool hi  = lane >= 16;
  const int nGroups = E_ >> 4;
  const int stride  = gridDim.x * WPB;

  float biasA[8], biasB[8];
#pragma unroll
  for (int nt = 0; nt < 8; nt++) {
    biasA[nt] = b_s2d[nt * 16 + lm];
    biasB[nt] = b_d2s[nt * 16 + lm];
  }

  for (int g = blockIdx.x * WPB + wid; g < nGroups; g += stride) {
    const int e0 = g << 4;

    // ---- prefetch next group's 32 rows (before any atomic asm clobber) ----
    {
      int gn = g + stride;
      if (gn < nGroups) {
        int en = gn << 4;
        int sv = 0, dv = 0;
        if (lane < 16) {
          sv = (int)ei[en + lane];
          dv = (int)ei[E_ + en + lane];
        }
        sv = __shfl(sv, lm, 32);
        dv = __shfl(dv, lm, 32);
        prefetch_row(x + (long)(hi ? dv : sv) * 128);
      }
    }

    int s_i = 0, d_i = 0;
    if (lane < 16) {
      s_i = (int)ei[e0 + lane];
      d_i = (int)ei[E_ + e0 + lane];
    }
    const int srcm = __shfl(s_i, lm, 32);
    const int dstm = __shfl(d_i, lm, 32);
    const float* ps = x + (long)srcm * 128;
    const float* pd = x + (long)dstm * 128;

    // scatter row indices for this lane's C-fragment rows (hoisted: 8+8 shfls)
    int dj[8], sj[8];
#pragma unroll
    for (int r = 0; r < 8; r++) {
      int row = hi ? r + 8 : r;
      dj[r] = __shfl(d_i, row, 32);
      sj[r] = __shfl(s_i, row, 32);
    }
    float s8[8];
#pragma unroll
    for (int r = 0; r < 8; r++) s8[r] = scores[e0 + (hi ? r + 8 : r)];

    // issue all gathers for BOTH GEMMs before any atomic (asm memory clobber)
    v16h afs[4], afd[4];
#pragma unroll
    for (int kt = 0; kt < 4; kt++) afs[kt] = make_a_frag(ps, kt * 32, hiOff, 1.0f);
#pragma unroll
    for (int kt = 0; kt < 4; kt++) afd[kt] = make_a_frag(pd, kt * 32, hiOff, 1.0f);

    // ---- msg_in = (x_src @ w_s2d + b) * score  -> h_in[dst] ----
    {
      v8f acc[8];
#pragma unroll
      for (int t = 0; t < 8; t++) { v8f z = {}; acc[t] = z; }
#pragma unroll
      for (int kt = 0; kt < 4; kt++)
#pragma unroll
        for (int nt = 0; nt < 8; nt++)
          acc[nt] = WMMA_F16(afs[kt], load_b_frag(lwA, kt * 8 + nt, lane), acc[nt]);
#pragma unroll
      for (int nt = 0; nt < 8; nt++) {
        int n = nt * 16 + lm;
#pragma unroll
        for (int r = 0; r < 8; r++)
          atomic_add_f32(h_in + (long)dj[r] * 128 + n,
                         (acc[nt][r] + biasA[nt]) * s8[r]);
      }
    }
    // ---- msg_out = (x_dst @ w_d2s + b) * score -> h_out[src] ----
    {
      v8f acc[8];
#pragma unroll
      for (int t = 0; t < 8; t++) { v8f z = {}; acc[t] = z; }
#pragma unroll
      for (int kt = 0; kt < 4; kt++)
#pragma unroll
        for (int nt = 0; nt < 8; nt++)
          acc[nt] = WMMA_F16(afd[kt], load_b_frag(lwB, kt * 8 + nt, lane), acc[nt]);
#pragma unroll
      for (int nt = 0; nt < 8; nt++) {
        int n = nt * 16 + lm;
#pragma unroll
        for (int r = 0; r < 8; r++)
          atomic_add_f32(h_out + (long)sj[r] * 128 + n,
                         (acc[nt][r] + biasB[nt]) * s8[r]);
      }
    }
  }
}

// ---------------- kernel 3: gate MLP + fuse + residual ----------------------
__global__ __launch_bounds__(256)
void node_kernel(const float* __restrict__ x,
                 const float* __restrict__ w_g1, const float* __restrict__ b_g1,
                 const float* __restrict__ w_g2, const float* __restrict__ b_g2,
                 const float* __restrict__ h_in, const float* __restrict__ h_out,
                 const float* __restrict__ in_deg, const float* __restrict__ out_deg,
                 float* __restrict__ out, int N_) {
  __shared__ __align__(32) _Float16 lw[8 * 8 * 32 * 16];   // 64KB: w_g1 (256x128)
  stage_w(w_g1, lw, 8);
  __syncthreads();

  const int lane = threadIdx.x & 31;
  const int wid  = threadIdx.x >> 5;
  const int lm   = lane & 15;
  const int hiOff = (lane & 16) >> 1;
  const bool hi  = lane >= 16;
  const int nGroups = N_ >> 4;
  const int stride  = gridDim.x * WPB;
  const float bg2 = b_g2[0];

  float biasv[8], w2v[8];
#pragma unroll
  for (int nt = 0; nt < 8; nt++) {
    biasv[nt] = b_g1[nt * 16 + lm];
    w2v[nt]   = w_g2[nt * 16 + lm];
  }

  for (int g = blockIdx.x * WPB + wid; g < nGroups; g += stride) {
    const int node = (g << 4) + lm;

    // prefetch next group's h_in/h_out rows
    {
      int gn = g + stride;
      if (gn < nGroups) {
        long nn = (long)((gn << 4) + lm) * 128;
        prefetch_row(hi ? (h_out + nn) : (h_in + nn));
      }
    }

    float idg = in_deg[node];  idg = idg < 1.0f ? 1.0f : idg;
    float odg = out_deg[node]; odg = odg < 1.0f ? 1.0f : odg;
    const float inv_i = 1.0f / idg;
    const float inv_o = 1.0f / odg;
    const float* pi = h_in  + (long)node * 128;
    const float* po = h_out + (long)node * 128;

    // all A-fragment loads up front: [h_in_norm | h_out_norm]
    v16h af[8];
#pragma unroll
    for (int kt = 0; kt < 4; kt++) af[kt]     = make_a_frag(pi, kt * 32, hiOff, inv_i);
#pragma unroll
    for (int kt = 0; kt < 4; kt++) af[kt + 4] = make_a_frag(po, kt * 32, hiOff, inv_o);

    v8f acc[8];
#pragma unroll
    for (int t = 0; t < 8; t++) { v8f z = {}; acc[t] = z; }

#pragma unroll
    for (int kt = 0; kt < 8; kt++)
#pragma unroll
      for (int nt = 0; nt < 8; nt++)
        acc[nt] = WMMA_F16(af[kt], load_b_frag(lw, kt * 8 + nt, lane), acc[nt]);

    float part[8];
#pragma unroll
    for (int r = 0; r < 8; r++) part[r] = 0.0f;
#pragma unroll
    for (int nt = 0; nt < 8; nt++)
#pragma unroll
      for (int r = 0; r < 8; r++) {
        float h = acc[nt][r] + biasv[nt];
        h = h > 0.0f ? h : 0.0f;
        part[r] += h * w2v[nt];
      }
#pragma unroll
    for (int off = 1; off < 16; off <<= 1)
#pragma unroll
      for (int r = 0; r < 8; r++) part[r] += __shfl_xor(part[r], off, 32);

    float gv[8];
#pragma unroll
    for (int r = 0; r < 8; r++) gv[r] = sigmoidf(part[r] + bg2);

    // broadcast g for this lane's own row (lane 0 holds rows 0-7, lane 16 rows 8-15)
    float gme = 0.0f;
#pragma unroll
    for (int r = 0; r < 8; r++) {
      float glo = __shfl(gv[r], 0, 32);
      float ghi = __shfl(gv[r], 16, 32);
      gme = (lm == r)     ? glo : gme;
      gme = (lm == r + 8) ? ghi : gme;
    }

    // fused output: lane covers row `lm`, columns [hi?64:0, +64)
    const int cbase = hi ? 64 : 0;
    const float* xr = x + (long)node * 128;
    float* orow = out + (long)node * 128;
    const float gm1 = 1.0f - gme;
#pragma unroll
    for (int c4 = 0; c4 < 16; c4++) {
      int c = cbase + c4 * 4;
      f4 a = *(const f4*)(pi + c);
      f4 b = *(const f4*)(po + c);
      f4 xx = *(const f4*)(xr + c);
      f4 o;
      o.x = gme * (a.x * inv_i) + gm1 * (b.x * inv_o) + xx.x;
      o.y = gme * (a.y * inv_i) + gm1 * (b.y * inv_o) + xx.y;
      o.z = gme * (a.z * inv_i) + gm1 * (b.z * inv_o) + xx.z;
      o.w = gme * (a.w * inv_i) + gm1 * (b.w * inv_o) + xx.w;
      *(f4*)(orow + c) = o;
    }
  }
}

// ---------------------------------------------------------------------------
extern "C" void kernel_launch(void* const* d_in, const int* in_sizes, int n_in,
                              void* d_out, int out_size, void* d_ws, size_t ws_size,
                              hipStream_t stream) {
  const float*     x     = (const float*)d_in[0];
  const long long* ei    = (const long long*)d_in[1];   // int64 per reference
  const float*     w_s2d = (const float*)d_in[2];
  const float*     b_s2d = (const float*)d_in[3];
  const float*     w_d2s = (const float*)d_in[4];
  const float*     b_d2s = (const float*)d_in[5];
  const float*     w_e1  = (const float*)d_in[6];
  const float*     b_e1  = (const float*)d_in[7];
  const float*     w_e2  = (const float*)d_in[8];
  const float*     b_e2  = (const float*)d_in[9];
  const float*     w_g1  = (const float*)d_in[10];
  const float*     b_g1  = (const float*)d_in[11];
  const float*     w_g2  = (const float*)d_in[12];
  const float*     b_g2  = (const float*)d_in[13];
  float* out = (float*)d_out;

  const int N_ = in_sizes[0] / 128;
  const int E_ = in_sizes[1] / 2;

  float* ws      = (float*)d_ws;
  float* h_in    = ws;
  float* h_out   = h_in  + (long)N_ * 128;
  float* in_deg  = h_out + (long)N_ * 128;
  float* out_deg = in_deg + N_;
  float* scores  = out_deg + N_;

  // zero h_in, h_out, in_deg, out_deg (contiguous; count divisible by 4)
  long zcount = (long)N_ * 128 * 2 + 2L * N_;
  zero_kernel<<<2048, 256, 0, stream>>>(ws, zcount / 4);

  edge_score_kernel<<<512, 256, 0, stream>>>(x, ei, w_e1, b_e1, w_e2, b_e2,
                                             scores, in_deg, out_deg, E_);
  edge_msg_kernel<<<512, 256, 0, stream>>>(x, ei, w_s2d, b_s2d, w_d2s, b_d2s,
                                           scores, h_in, h_out, E_);
  node_kernel<<<128, 256, 0, stream>>>(x, w_g1, b_g1, w_g2, b_g2,
                                       h_in, h_out, in_deg, out_deg, out, N_);
}
